// TransformerEncoderLayer_58153857187988
// MI455X (gfx1250) — matrix operand order repaired and checked by reading, not compiled, not run on hardware
//
#include <hip/hip_runtime.h>
#include <hip/hip_bf16.h>
#include <math.h>

typedef __attribute__((ext_vector_type(8)))  _Float16 v8h;
typedef __attribute__((ext_vector_type(16))) _Float16 v16h;
typedef __attribute__((ext_vector_type(8)))  float    v8f;

#define BB   8
#define SS   1024
#define DD   768
#define HH   12
#define DHD  64
#define FF   3072
#define BSR  (BB * SS)          // 8192 rows
#define LN_EPS 1e-5f

// ---------------------------------------------------------------------------
// helpers
// ---------------------------------------------------------------------------
static __device__ __forceinline__ v16h ld2(const _Float16* p0, const _Float16* p1) {
  v8h a = *(const v8h*)p0;
  v8h b = *(const v8h*)p1;
  v16h r;
#pragma unroll
  for (int i = 0; i < 8; ++i) { r[i] = a[i]; r[i + 8] = b[i]; }
  return r;
}

static __device__ __forceinline__ v8f wmma32(v16h a, v16h b, v8f c) {
  return __builtin_amdgcn_wmma_f32_16x16x32_f16(false, a, false, b, (short)0, c,
                                                false, false);
}

// ---------------------------------------------------------------------------
// weight pack: W[kdim x ndim] f32 row-major  ->  f16 blocked B-operand layout
// block (kt,nt) holds 32 lanes * 16 halves; lane l: col nt*16+(l&15),
// K = kt*32 + (l>>4)*16 + i  (i=0..15, contiguous per lane)
// ---------------------------------------------------------------------------
__global__ __launch_bounds__(256) void pack_b_kernel(
    const float* __restrict__ src, _Float16* __restrict__ dst,
    int kdim, int ndim, int count) {
  size_t per = (size_t)kdim * ndim;
  size_t tid = (size_t)blockIdx.x * 256 + threadIdx.x;
  if (tid >= per * (size_t)count) return;
  int    mat = (int)(tid / per);
  size_t p   = tid % per;
  int i    = (int)(p & 15);
  int lane = (int)((p >> 4) & 31);
  size_t blk = p >> 9;
  int NT = ndim >> 4;
  int nt = (int)(blk % NT);
  int kt = (int)(blk / NT);
  int n = nt * 16 + (lane & 15);
  int k = kt * 32 + ((lane >> 4) << 4) + i;
  dst[mat * per + p] = (_Float16)src[mat * per + (size_t)k * ndim + n];
}

// ---------------------------------------------------------------------------
// LayerNorm: f32 [rows x 768] -> f16 row-major
// ---------------------------------------------------------------------------
__global__ __launch_bounds__(256) void ln_kernel(
    const float* __restrict__ x, const float* __restrict__ w,
    const float* __restrict__ bv, _Float16* __restrict__ out) {
  __shared__ float red[8];
  int row = blockIdx.x;
  const float* xr = x + (size_t)row * DD;
  int t = threadIdx.x;
  float v0 = xr[t], v1 = xr[t + 256], v2 = xr[t + 512];
  float s = v0 + v1 + v2;
#pragma unroll
  for (int m = 16; m; m >>= 1) s += __shfl_xor(s, m, 32);
  if ((t & 31) == 0) red[t >> 5] = s;
  __syncthreads();
  float tot = 0.f;
#pragma unroll
  for (int i = 0; i < 8; ++i) tot += red[i];
  float mean = tot * (1.0f / DD);
  float d0 = v0 - mean, d1 = v1 - mean, d2 = v2 - mean;
  float s2 = d0 * d0 + d1 * d1 + d2 * d2;
#pragma unroll
  for (int m = 16; m; m >>= 1) s2 += __shfl_xor(s2, m, 32);
  __syncthreads();
  if ((t & 31) == 0) red[t >> 5] = s2;
  __syncthreads();
  float tv = 0.f;
#pragma unroll
  for (int i = 0; i < 8; ++i) tv += red[i];
  float rs = rsqrtf(tv * (1.0f / DD) + LN_EPS);
  _Float16* o = out + (size_t)row * DD;
  o[t]       = (_Float16)(d0 * rs * w[t]       + bv[t]);
  o[t + 256] = (_Float16)(d1 * rs * w[t + 256] + bv[t + 256]);
  o[t + 512] = (_Float16)(d2 * rs * w[t + 512] + bv[t + 512]);
}

// ---------------------------------------------------------------------------
// QKV projection: per (b,h) GEMM  [1024 x 64] = h1[b] (1024x768) * Wpacked[h]
// wave tile = 16x64, copy-free ping-pong double buffering over K (KT even).
// ---------------------------------------------------------------------------
__global__ __launch_bounds__(256) void gemm_qkv_kernel(
    const _Float16* __restrict__ A, const _Float16* __restrict__ Bp,
    const float* __restrict__ bias, _Float16* __restrict__ out,
    float scale, int transposed) {
  int wid = threadIdx.x >> 5, lane = threadIdx.x & 31;
  int w = blockIdx.x * 8 + wid;
  int mtile = w & 63;
  int bh = w >> 6;
  int h = bh % HH;
  int ll = lane & 15, sel = lane >> 4;
  int m0 = mtile * 16;

  const _Float16* Ar  = A + (size_t)(bh / HH) * SS * DD + (size_t)(m0 + ll) * DD;
  const _Float16* Bph = Bp + (size_t)h * DD * DHD;
  const float*    bi  = bias + h * DHD;

  v8f acc[4];
#pragma unroll
  for (int j = 0; j < 4; ++j) acc[j] = (v8f){};

  const int KT = DD / 32;  // 24 (even)
  v16h aA, aB, bA[4], bB[4];

#define QKV_LOAD(aR, bR, ktv)                                              \
  do {                                                                     \
    const _Float16* ap_ = Ar + (size_t)(ktv) * 32 + sel * 16;              \
    aR = ld2(ap_, ap_ + 8);                                                \
    _Pragma("unroll") for (int j_ = 0; j_ < 4; ++j_) {                     \
      const _Float16* bp_ = Bph + (size_t)((ktv) * 4 + j_) * 512 + lane * 16; \
      bR[j_] = ld2(bp_, bp_ + 8);                                          \
    }                                                                      \
  } while (0)

  QKV_LOAD(aA, bA, 0);
  for (int kt = 0; kt < KT - 2; kt += 2) {
    QKV_LOAD(aB, bB, kt + 1);
#pragma unroll
    for (int j = 0; j < 4; ++j) acc[j] = wmma32(aA, bA[j], acc[j]);
    QKV_LOAD(aA, bA, kt + 2);
#pragma unroll
    for (int j = 0; j < 4; ++j) acc[j] = wmma32(aB, bB[j], acc[j]);
  }
  QKV_LOAD(aB, bB, KT - 1);
#pragma unroll
  for (int j = 0; j < 4; ++j) acc[j] = wmma32(aA, bA[j], acc[j]);
#pragma unroll
  for (int j = 0; j < 4; ++j) acc[j] = wmma32(aB, bB[j], acc[j]);
#undef QKV_LOAD

#pragma unroll
  for (int j = 0; j < 4; ++j) {
    int n = j * 16 + ll;
    float bb = bi[n];
#pragma unroll
    for (int i = 0; i < 8; ++i) {
      int row = m0 + sel * 8 + i;
      float val = (acc[j][i] + bb) * scale;
      if (transposed)
        out[(size_t)bh * SS * DHD + (size_t)n * SS + row] = (_Float16)val;
      else
        out[(size_t)bh * SS * DHD + (size_t)row * DHD + n] = (_Float16)val;
    }
  }
}

// ---------------------------------------------------------------------------
// Flash attention: per wave one 16-row Q tile of one (b,h).
// q,k: [bh][1024][64] f16 (q pre-scaled by 1/8);  vt: [bh][64][1024] f16
// ctx out: [b][s][h*64+e] f16 row-major (feeds Wo GEMM as A operand)
// ---------------------------------------------------------------------------
__global__ __launch_bounds__(256) void attn_kernel(
    const _Float16* __restrict__ q, const _Float16* __restrict__ k,
    const _Float16* __restrict__ vt, _Float16* __restrict__ ctx) {
  __shared__ _Float16 plds[8 * 512];
  int wid = threadIdx.x >> 5, lane = threadIdx.x & 31;
  int w = blockIdx.x * 8 + wid;
  int mtile = w & 63;
  int bh = w >> 6;
  int b = bh / HH, h = bh % HH;
  int ll = lane & 15, sel = lane >> 4;
  int m0 = mtile * 16;

  const _Float16* qb = q  + (size_t)bh * SS * DHD;
  const _Float16* kb = k  + (size_t)bh * SS * DHD;
  const _Float16* vb = vt + (size_t)bh * SS * DHD;
  _Float16* pw = plds + wid * 512;

  const _Float16* qr = qb + (size_t)(m0 + ll) * DHD;
  v16h a0 = ld2(qr + sel * 16,      qr + 8 + sel * 16);
  v16h a1 = ld2(qr + 32 + sel * 16, qr + 40 + sel * 16);

  v8f o[4];
#pragma unroll
  for (int j = 0; j < 4; ++j) o[j] = (v8f){};
  float mrun[8], lrun[8];
#pragma unroll
  for (int i = 0; i < 8; ++i) { mrun[i] = -3.0e38f; lrun[i] = 0.f; }

  for (int t0 = 0; t0 < SS; t0 += 32) {
    // ---- all 4 score B tiles up front (overlap loads with WMMA issue) ----
    v16h kb4[4];
#pragma unroll
    for (int j = 0; j < 2; ++j)
#pragma unroll
      for (int kt = 0; kt < 2; ++kt) {
        const _Float16* bp =
            kb + (size_t)(t0 + j * 16 + ll) * DHD + kt * 32 + sel * 16;
        kb4[j * 2 + kt] = ld2(bp, bp + 8);
      }
    v8f s[2];
    s[0] = (v8f){}; s[1] = (v8f){};
    s[0] = wmma32(a0, kb4[0], s[0]);
    s[0] = wmma32(a1, kb4[1], s[0]);
    s[1] = wmma32(a0, kb4[2], s[1]);
    s[1] = wmma32(a1, kb4[3], s[1]);

    // ---- V tiles issued early: independent of softmax math below ----
    v16h vb4[4];
#pragma unroll
    for (int j = 0; j < 4; ++j) {
      const _Float16* vp = vb + (size_t)(j * 16 + ll) * SS + t0 + sel * 16;
      vb4[j] = ld2(vp, vp + 8);
    }
    // prefetch next iteration's K/V tiles (wraps on last iter -> harmless)
    {
      int tn = (t0 + 32) & (SS - 1);
      __builtin_prefetch(kb + (size_t)(tn + ll) * DHD, 0, 3);
      __builtin_prefetch(kb + (size_t)(tn + 16 + ll) * DHD, 0, 3);
      __builtin_prefetch(vb + (size_t)(sel * 32 + ll) * SS + tn, 0, 3);
    }

    // ---- online softmax per row (row = half-wave x vgpr index) ----
#pragma unroll
    for (int i = 0; i < 8; ++i) {
      float mt = fmaxf(s[0][i], s[1][i]);
#pragma unroll
      for (int m = 8; m; m >>= 1) mt = fmaxf(mt, __shfl_xor(mt, m, 32));
      float mn = fmaxf(mrun[i], mt);
      float alpha = __expf(mrun[i] - mn);
      mrun[i] = mn;
      float p0 = __expf(s[0][i] - mn);
      float p1 = __expf(s[1][i] - mn);
      s[0][i] = p0; s[1][i] = p1;
      float rs = p0 + p1;
#pragma unroll
      for (int m = 8; m; m >>= 1) rs += __shfl_xor(rs, m, 32);
      lrun[i] = lrun[i] * alpha + rs;
#pragma unroll
      for (int j = 0; j < 4; ++j) o[j][i] *= alpha;
    }
    // ---- P (C layout f32) -> LDS row-major 16x32 f16 ----
#pragma unroll
    for (int j = 0; j < 2; ++j)
#pragma unroll
      for (int i = 0; i < 8; ++i)
        pw[(sel * 8 + i) * 32 + j * 16 + ll] = (_Float16)s[j][i];
    // read back as A operand (DS ops in-order within wave)
    const _Float16* pr = pw + ll * 32 + sel * 16;
    v16h pa = ld2(pr, pr + 8);
    // ---- O += P * V ----
#pragma unroll
    for (int j = 0; j < 4; ++j) o[j] = wmma32(pa, vb4[j], o[j]);
  }
#pragma unroll
  for (int i = 0; i < 8; ++i) {
    float inv = 1.0f / lrun[i];
    int row = m0 + sel * 8 + i;
#pragma unroll
    for (int j = 0; j < 4; ++j) {
      int col = h * DHD + j * 16 + ll;
      ctx[((size_t)b * SS + row) * DD + col] = (_Float16)(o[j][i] * inv);
    }
  }
}

// ---------------------------------------------------------------------------
// Generic GEMM with epilogue: C[M x N] = A[M x K](f16) * Bpacked + bias
// optional exact GELU, optional residual (f32, stride N), f32 and/or f16 out.
// wave tile 16x64, copy-free ping-pong double buffering over K (KT even).
// ---------------------------------------------------------------------------
__global__ __launch_bounds__(256) void gemm_ep_kernel(
    const _Float16* __restrict__ A, const _Float16* __restrict__ Bp,
    const float* __restrict__ bias, float* __restrict__ outf,
    _Float16* __restrict__ outh, const float* __restrict__ resid,
    int M, int N, int K, int gelu) {
  int wid = threadIdx.x >> 5, lane = threadIdx.x & 31;
  int w = blockIdx.x * 8 + wid;
  int nb4 = N >> 6;
  int mtile = w / nb4, nb = w % nb4;
  int m0 = mtile * 16, n0 = nb * 64;
  int NT = N >> 4;
  int nblk0 = n0 >> 4;
  int ll = lane & 15, sel = lane >> 4;

  const _Float16* Ar = A + (size_t)(m0 + ll) * K;
  v8f acc[4];
#pragma unroll
  for (int j = 0; j < 4; ++j) acc[j] = (v8f){};

  const int KT = K >> 5;  // 24 or 96 (even)
  v16h aA, aB, bA[4], bB[4];

#define EP_LOAD(aR, bR, ktv)                                                  \
  do {                                                                        \
    const _Float16* ap_ = Ar + (size_t)(ktv) * 32 + sel * 16;                 \
    aR = ld2(ap_, ap_ + 8);                                                   \
    _Pragma("unroll") for (int j_ = 0; j_ < 4; ++j_) {                        \
      const _Float16* bp_ =                                                   \
          Bp + ((size_t)(ktv) * NT + nblk0 + j_) * 512 + lane * 16;           \
      bR[j_] = ld2(bp_, bp_ + 8);                                             \
    }                                                                         \
  } while (0)

  EP_LOAD(aA, bA, 0);
  for (int kt = 0; kt < KT - 2; kt += 2) {
    EP_LOAD(aB, bB, kt + 1);
#pragma unroll
    for (int j = 0; j < 4; ++j) acc[j] = wmma32(aA, bA[j], acc[j]);
    EP_LOAD(aA, bA, kt + 2);
#pragma unroll
    for (int j = 0; j < 4; ++j) acc[j] = wmma32(aB, bB[j], acc[j]);
  }
  EP_LOAD(aB, bB, KT - 1);
#pragma unroll
  for (int j = 0; j < 4; ++j) acc[j] = wmma32(aA, bA[j], acc[j]);
#pragma unroll
  for (int j = 0; j < 4; ++j) acc[j] = wmma32(aB, bB[j], acc[j]);
#undef EP_LOAD

#pragma unroll
  for (int j = 0; j < 4; ++j) {
    int n = n0 + j * 16 + ll;
    float bb = bias[n];
#pragma unroll
    for (int i = 0; i < 8; ++i) {
      size_t row = (size_t)(m0 + sel * 8 + i);
      float val = acc[j][i] + bb;
      if (gelu) val = 0.5f * val * (1.0f + erff(val * 0.70710678118f));
      if (resid) val += resid[row * N + n];
      if (outf) outf[row * N + n] = val;
      if (outh) outh[row * N + n] = (_Float16)val;
    }
  }
}

// ---------------------------------------------------------------------------
// host side
// ---------------------------------------------------------------------------
extern "C" void kernel_launch(void* const* d_in, const int* in_sizes, int n_in,
                              void* d_out, int out_size, void* d_ws,
                              size_t ws_size, hipStream_t stream) {
  (void)in_sizes; (void)n_in; (void)out_size; (void)ws_size;
  const float* x    = (const float*)d_in[0];
  const float* ln1w = (const float*)d_in[1];
  const float* ln1b = (const float*)d_in[2];
  const float* ln2w = (const float*)d_in[3];
  const float* ln2b = (const float*)d_in[4];
  const float* Wq   = (const float*)d_in[5];
  const float* bq   = (const float*)d_in[6];
  const float* Wk   = (const float*)d_in[7];
  const float* bk   = (const float*)d_in[8];
  const float* Wv   = (const float*)d_in[9];
  const float* bv   = (const float*)d_in[10];
  const float* Wo   = (const float*)d_in[11];
  const float* bo   = (const float*)d_in[12];
  const float* f1w  = (const float*)d_in[13];
  const float* f1b  = (const float*)d_in[14];
  const float* f2w  = (const float*)d_in[15];
  const float* f2b  = (const float*)d_in[16];
  float* out = (float*)d_out;

  char* ws = (char*)d_ws;
  size_t off = 0;
  auto take = [&](size_t bytes) {
    size_t o = off;
    off += (bytes + 255) & ~(size_t)255;
    return o;
  };
  const size_t ACT16 = (size_t)BSR * DD * 2;        // 12.6 MB
  size_t o_h1  = take(ACT16);                       // LN1 out f16 (reused by LN2)
  size_t o_q   = take(ACT16);                       // q  (reused by FFN hidden)
  size_t o_k   = take(ACT16);
  size_t o_vt  = take(ACT16);
  size_t o_ctx = take(ACT16);
  size_t o_x1  = take((size_t)BSR * DD * 4);        // post-attn residual f32
  size_t o_pq  = take((size_t)DD * DD * 2);         // packed Wq (12 heads)
  size_t o_pk  = take((size_t)DD * DD * 2);
  size_t o_pv  = take((size_t)DD * DD * 2);
  size_t o_po  = take((size_t)DD * DD * 2);
  size_t o_p1  = take((size_t)DD * FF * 2);
  size_t o_p2  = take((size_t)FF * DD * 2);

  _Float16* h1  = (_Float16*)(ws + o_h1);
  _Float16* qh  = (_Float16*)(ws + o_q);
  _Float16* kh  = (_Float16*)(ws + o_k);
  _Float16* vth = (_Float16*)(ws + o_vt);
  _Float16* ctx = (_Float16*)(ws + o_ctx);
  float*    x1  = (float*)(ws + o_x1);
  _Float16* pQ  = (_Float16*)(ws + o_pq);
  _Float16* pK  = (_Float16*)(ws + o_pk);
  _Float16* pV  = (_Float16*)(ws + o_pv);
  _Float16* pO  = (_Float16*)(ws + o_po);
  _Float16* pF1 = (_Float16*)(ws + o_p1);
  _Float16* pF2 = (_Float16*)(ws + o_p2);
  _Float16* hf  = (_Float16*)(ws + o_q);   // FFN hidden f16, reuses q..ctx (50.3MB)
  _Float16* h2  = h1;                      // LN2 out reuses LN1 buffer

  auto packN = [&](const float* src, _Float16* dst, int kd, int nd, int cnt) {
    size_t tot = (size_t)kd * nd * cnt;
    pack_b_kernel<<<(unsigned)((tot + 255) / 256), 256, 0, stream>>>(src, dst, kd, nd, cnt);
  };
  packN(Wq, pQ, DD, DHD, HH);
  packN(Wk, pK, DD, DHD, HH);
  packN(Wv, pV, DD, DHD, HH);
  packN(Wo, pO, DD, DD, 1);
  packN(f1w, pF1, DD, FF, 1);
  packN(f2w, pF2, FF, DD, 1);

  // LN1
  ln_kernel<<<BSR, 256, 0, stream>>>(x, ln1w, ln1b, h1);

  // QKV projections (96 bh-GEMMs each, 8 waves/block)
  const int qkvBlocks = BB * HH * (SS / 16) / 8;    // 768
  gemm_qkv_kernel<<<qkvBlocks, 256, 0, stream>>>(h1, pQ, bq, qh, 0.125f, 0);
  gemm_qkv_kernel<<<qkvBlocks, 256, 0, stream>>>(h1, pK, bk, kh, 1.0f, 0);
  gemm_qkv_kernel<<<qkvBlocks, 256, 0, stream>>>(h1, pV, bv, vth, 1.0f, 1);

  // attention
  attn_kernel<<<qkvBlocks, 256, 0, stream>>>(qh, kh, vth, ctx);

  // output projection + residual -> x1 (f32)
  {
    int waves = (BSR / 16) * (DD / 64);             // 6144
    gemm_ep_kernel<<<waves / 8, 256, 0, stream>>>(ctx, pO, bo, x1, nullptr, x,
                                                  BSR, DD, DD, 0);
  }
  // LN2
  ln_kernel<<<BSR, 256, 0, stream>>>(x1, ln2w, ln2b, h2);
  // FC1 + exact GELU -> hf (f16)
  {
    int waves = (BSR / 16) * (FF / 64);             // 24576
    gemm_ep_kernel<<<waves / 8, 256, 0, stream>>>(h2, pF1, f1b, nullptr, hf,
                                                  nullptr, BSR, FF, DD, 1);
  }
  // FC2 + residual -> d_out (f32)
  {
    int waves = (BSR / 16) * (DD / 64);             // 6144
    gemm_ep_kernel<<<waves / 8, 256, 0, stream>>>(hf, pF2, f2b, out, nullptr,
                                                  x1, BSR, DD, FF, 0);
  }
}
